// resolution_attention_43825846289084
// MI455X (gfx1250) — matrix operand order repaired and checked by reading, not compiled, not run on hardware
//
#include <hip/hip_runtime.h>

typedef __attribute__((ext_vector_type(16))) _Float16 v16h;
typedef __attribute__((ext_vector_type(8)))  _Float16 v8h;
typedef __attribute__((ext_vector_type(8)))  float    v8f;
typedef __attribute__((ext_vector_type(4)))  float    v4f;

#define DIM 768
#define NQ 128            // 4 groups * 32 queries
#define KEYS 784
#define RSQRT_D 0.03608439182435161f   // 1/sqrt(768)

// ---------------------------------------------------------------------------
// Pack an f32 weight matrix into f16 WMMA B-fragments.
// Fragment f = kt*(N/16)+nt holds a 32x16 B tile; within it, lane l supplies
// column (l&15) and K-rows (l<16 ? kt*32+0..7,16..23 : kt*32+8..15,24..31) in halves 0..15.
// src element: transpose ? W[n*K+k] : W[k*N+n]
// ---------------------------------------------------------------------------
__global__ void pack_b_kernel(const float* __restrict__ W, _Float16* __restrict__ Bp,
                              int K, int N, int transpose) {
    int idx = blockIdx.x * blockDim.x + threadIdx.x;
    if (idx >= K * N) return;
    int h    = idx & 15;
    int lane = (idx >> 4) & 31;
    int f    = idx >> 9;
    int ntiles = N >> 4;
    int nt = f % ntiles, kt = f / ntiles;
    int k = kt * 32 + ((h & 8) << 1) + ((lane & 16) ? 8 : 0) + (h & 7);
    int n = nt * 16 + (lane & 15);
    float s = transpose ? W[(size_t)n * K + k] : W[(size_t)k * N + n];
    Bp[idx] = (_Float16)s;
}

__global__ void f32_to_f16_kernel(const float* __restrict__ src,
                                  _Float16* __restrict__ dst, int n) {
    int i = blockIdx.x * blockDim.x + threadIdx.x;
    if (i < n) dst[i] = (_Float16)src[i];
}

// ---------------------------------------------------------------------------
// C[M,N] = A(f16 row-major, MxK) * Bp(packed f16) [+ bias[N]] [+ res[M,N]]
// Writes f32 C (if Cf) and/or f16 C (if Ch).  Block = 8 waves; each wave owns
// one 16x16 D tile; block tile = 16(M) x 128(N).
// ---------------------------------------------------------------------------
__global__ __launch_bounds__(256)
void wmma_gemm_kernel(const _Float16* __restrict__ A, const _Float16* __restrict__ Bp,
                      const float* __restrict__ bias, const float* __restrict__ res,
                      float* __restrict__ Cf, _Float16* __restrict__ Ch,
                      int M, int N, int K) {
    const int lane = threadIdx.x & 31;
    const int wave = threadIdx.x >> 5;
    const int m0   = blockIdx.x * 16;
    const int nt   = blockIdx.y * 8 + wave;      // 16-wide N tile index
    const int ntiles = N >> 4;
    const int col  = lane & 15;
    const int hi   = (lane & 16) ? 1 : 0;

    v8f c;
    float b = bias ? bias[nt * 16 + col] : 0.0f;
    for (int i = 0; i < 8; ++i) c[i] = b;

    const _Float16* arow = A + (size_t)(m0 + col) * K + hi * 8;
    for (int kt = 0; kt < (K >> 5); ++kt) {
        // A fragment: lane<16 -> K 0..7 & 16..23 of this 32-slab; lane>=16 -> 8..15 & 24..31
        v8h lo = *(const v8h*)(arow + kt * 32);
        v8h hp = *(const v8h*)(arow + kt * 32 + 16);
        v16h a;
        for (int i = 0; i < 8; ++i) { a[i] = lo[i]; a[i + 8] = hp[i]; }
        // B fragment: one contiguous 32B per lane (pre-packed)
        v16h bf = ((const v16h*)Bp)[((size_t)(kt * ntiles + nt)) * 32 + lane];
        c = __builtin_amdgcn_wmma_f32_16x16x32_f16(false, a, false, bf,
                                                   (short)0, c, false, false);
    }

    const int n = nt * 16 + col;
    for (int i = 0; i < 8; ++i) {
        int m = m0 + hi * 8 + i;               // D layout: lanes>=16 hold rows M=8..15
        size_t o = (size_t)m * N + n;
        float v = c[i];
        if (res) v += res[o];
        if (Cf)  Cf[o] = v;
        if (Ch)  Ch[o] = (_Float16)v;
    }
}

// ---------------------------------------------------------------------------
// Streaming flash pass: block b handles query q=b>>1, key half h=b&1
// (keys [h*392, h*392+392)).  Per key: score = qk . hr_row * rsqrt(D),
// online softmax with the row kept in registers for the weighted accumulate.
// HR is read exactly once, non-temporally.  Emits partial (m, l, acc).
// ---------------------------------------------------------------------------
__global__ __launch_bounds__(512)
void attn_stream_kernel(const float* __restrict__ QK, const float* __restrict__ HR,
                        float* __restrict__ Pm, float* __restrict__ Pl,
                        float* __restrict__ Pacc) {
    const int blk  = blockIdx.x;               // 0..255
    const int q    = blk >> 1;
    const int half = blk & 1;
    const int lane = threadIdx.x & 31;
    const int wave = threadIdx.x >> 5;         // 0..15
    const size_t hr_base = (size_t)q * KEYS * DIM;

    const v4f* qkv = (const v4f*)(QK + (size_t)q * DIM);
    v4f qk[6];
    for (int c = 0; c < 6; ++c) qk[c] = qkv[c * 32 + lane];

    float m = -1e30f, l = 0.0f;
    v4f acc[6];
    for (int c = 0; c < 6; ++c) acc[c] = (v4f)(0.0f);

    const int j0 = half * 392, j1 = j0 + 392;
    for (int j = j0 + wave; j < j1; j += 16) {
        const v4f* row = (const v4f*)(HR + hr_base + (size_t)j * DIM);
        v4f r[6];
        for (int c = 0; c < 6; ++c)
            r[c] = __builtin_nontemporal_load(&row[c * 32 + lane]);
        if (j + 16 < j1)
            __builtin_prefetch(HR + hr_base + (size_t)(j + 16) * DIM + lane * 16, 0, 0);

        float p = 0.0f;
        for (int c = 0; c < 6; ++c) {
            v4f t = qk[c] * r[c];
            p += t[0] + t[1] + t[2] + t[3];
        }
        for (int off = 16; off > 0; off >>= 1) p += __shfl_xor(p, off, 32);
        float score = p * RSQRT_D;

        float nm    = fmaxf(m, score);
        float scale = __expf(m - nm);          // first iter: exp(-huge) == 0
        float w     = __expf(score - nm);
        l = l * scale + w;
        for (int c = 0; c < 6; ++c) acc[c] = acc[c] * scale + r[c] * w;
        m = nm;
    }

    __shared__ float sm[16], sl[16];
    __shared__ float sacc[DIM];
    for (int e = threadIdx.x; e < DIM; e += 512) sacc[e] = 0.0f;
    if (lane == 0) { sm[wave] = m; sl[wave] = l; }
    __syncthreads();

    float M = -1e30f;
    for (int i = 0; i < 16; ++i) M = fmaxf(M, sm[i]);
    float L = 0.0f;
    for (int i = 0; i < 16; ++i) L += sl[i] * __expf(sm[i] - M);

    float f = __expf(m - M);
    for (int c = 0; c < 6; ++c)
        for (int t = 0; t < 4; ++t)
            atomicAdd(&sacc[c * 128 + lane * 4 + t], acc[c][t] * f);   // ds_add_f32
    __syncthreads();

    if (threadIdx.x == 0) { Pm[blk] = M; Pl[blk] = L; }
    for (int e = threadIdx.x; e < DIM; e += 512)
        Pacc[(size_t)blk * DIM + e] = sacc[e];
}

// Merge the two key-half partials per query and normalize -> u (f16 for GEMM).
__global__ __launch_bounds__(256)
void attn_combine_kernel(const float* __restrict__ Pm, const float* __restrict__ Pl,
                         const float* __restrict__ Pacc, _Float16* __restrict__ Uh) {
    const int q = blockIdx.x;
    float m0 = Pm[2 * q], m1 = Pm[2 * q + 1];
    float M  = fmaxf(m0, m1);
    float f0 = __expf(m0 - M), f1 = __expf(m1 - M);
    float inv = 1.0f / (Pl[2 * q] * f0 + Pl[2 * q + 1] * f1);
    const float* a0 = Pacc + (size_t)(2 * q) * DIM;
    const float* a1 = a0 + DIM;
    for (int e = threadIdx.x; e < DIM; e += 256)
        Uh[(size_t)q * DIM + e] = (_Float16)((a0[e] * f0 + a1[e] * f1) * inv);
}

// Row LayerNorm over D=768.
__global__ __launch_bounds__(256)
void layernorm_kernel(const float* __restrict__ Y, const float* __restrict__ gamma,
                      const float* __restrict__ beta, float* __restrict__ out) {
    const int r = blockIdx.x;
    const float* y = Y + (size_t)r * DIM;
    const int lane = threadIdx.x & 31, wave = threadIdx.x >> 5;
    float s = 0.0f, s2 = 0.0f;
    for (int e = threadIdx.x; e < DIM; e += 256) { float v = y[e]; s += v; s2 += v * v; }
    for (int off = 16; off > 0; off >>= 1) {
        s  += __shfl_xor(s,  off, 32);
        s2 += __shfl_xor(s2, off, 32);
    }
    __shared__ float ss[8], ss2[8];
    if (lane == 0) { ss[wave] = s; ss2[wave] = s2; }
    __syncthreads();
    float ts = 0.0f, ts2 = 0.0f;
    for (int i = 0; i < 8; ++i) { ts += ss[i]; ts2 += ss2[i]; }
    float mu  = ts * (1.0f / DIM);
    float var = ts2 * (1.0f / DIM) - mu * mu;
    float rs  = rsqrtf(var + 1e-5f);
    for (int e = threadIdx.x; e < DIM; e += 256)
        out[(size_t)r * DIM + e] = (y[e] - mu) * rs * gamma[e] + beta[e];
}

// ---------------------------------------------------------------------------
extern "C" void kernel_launch(void* const* d_in, const int* in_sizes, int n_in,
                              void* d_out, int out_size, void* d_ws, size_t ws_size,
                              hipStream_t stream) {
    (void)in_sizes; (void)n_in; (void)out_size; (void)ws_size;
    const float* LR = (const float*)d_in[0];
    const float* HR = (const float*)d_in[1];
    const float* Wq = (const float*)d_in[2];
    const float* bq = (const float*)d_in[3];
    const float* Wk = (const float*)d_in[4];
    // d_in[5] = bk: q.bk is constant per query -> cancels in softmax, unused.
    const float* Wv = (const float*)d_in[6];
    const float* bv = (const float*)d_in[7];
    const float* Wo = (const float*)d_in[8];
    const float* bo = (const float*)d_in[9];
    const float* gamma = (const float*)d_in[10];
    const float* beta  = (const float*)d_in[11];

    char* ws = (char*)d_ws;
    size_t off = 0;
    auto alloc = [&](size_t bytes) -> void* {
        void* p = ws + off;
        off += (bytes + 255) & ~(size_t)255;
        return p;
    };
    _Float16* WqT  = (_Float16*)alloc((size_t)DIM * DIM * 2);  // Wq^T packed
    _Float16* WkP  = (_Float16*)alloc((size_t)DIM * DIM * 2);  // Wk   packed (no transpose)
    _Float16* WvT  = (_Float16*)alloc((size_t)DIM * DIM * 2);  // Wv^T packed
    _Float16* WoT  = (_Float16*)alloc((size_t)DIM * DIM * 2);  // Wo^T packed
    _Float16* LRh  = (_Float16*)alloc((size_t)NQ * DIM * 2);
    float*    Qf   = (float*)   alloc((size_t)NQ * DIM * 4);   // q (f32, residual)
    _Float16* Qh   = (_Float16*)alloc((size_t)NQ * DIM * 2);
    float*    QKf  = (float*)   alloc((size_t)NQ * DIM * 4);   // q @ Wk
    float*    Pm   = (float*)   alloc(256 * 4);
    float*    Pl   = (float*)   alloc(256 * 4);
    float*    Pacc = (float*)   alloc((size_t)256 * DIM * 4);
    _Float16* Uh   = (_Float16*)alloc((size_t)NQ * DIM * 2);   // softmax-weighted HR rows
    _Float16* AVh  = (_Float16*)alloc((size_t)NQ * DIM * 2);   // u @ Wv^T + bv
    float*    Yf   = (float*)   alloc((size_t)NQ * DIM * 4);   // q + attn @ Wo^T + bo

    const int pg = (DIM * DIM + 255) / 256;
    pack_b_kernel<<<pg, 256, 0, stream>>>(Wq, WqT, DIM, DIM, 1);
    pack_b_kernel<<<pg, 256, 0, stream>>>(Wk, WkP, DIM, DIM, 0);
    pack_b_kernel<<<pg, 256, 0, stream>>>(Wv, WvT, DIM, DIM, 1);
    pack_b_kernel<<<pg, 256, 0, stream>>>(Wo, WoT, DIM, DIM, 1);
    f32_to_f16_kernel<<<(NQ * DIM + 255) / 256, 256, 0, stream>>>(LR, LRh, NQ * DIM);

    dim3 gg(NQ / 16, DIM / 128);
    // q = LR @ Wq^T + bq
    wmma_gemm_kernel<<<gg, 256, 0, stream>>>(LRh, WqT, bq, nullptr, Qf, Qh, NQ, DIM, DIM);
    // qk = q @ Wk
    wmma_gemm_kernel<<<gg, 256, 0, stream>>>(Qh, WkP, nullptr, nullptr, QKf, nullptr, NQ, DIM, DIM);
    // streaming softmax over HR (read once, non-temporal)
    attn_stream_kernel<<<256, 512, 0, stream>>>(QKf, HR, Pm, Pl, Pacc);
    attn_combine_kernel<<<NQ, 256, 0, stream>>>(Pm, Pl, Pacc, Uh);
    // attn = u @ Wv^T + bv
    wmma_gemm_kernel<<<gg, 256, 0, stream>>>(Uh, WvT, bv, nullptr, nullptr, AVh, NQ, DIM, DIM);
    // y = q + attn @ Wo^T + bo
    wmma_gemm_kernel<<<gg, 256, 0, stream>>>(AVh, WoT, bo, Qf, Yf, nullptr, NQ, DIM, DIM);
    // out = layernorm(y) * gamma + beta
    layernorm_kernel<<<NQ, 256, 0, stream>>>(Yf, gamma, beta, (float*)d_out);
}
